// MultiHeadAttention_50517405335944
// MI455X (gfx1250) — matrix operand order repaired and checked by reading, not compiled, not run on hardware
//
#include <hip/hip_runtime.h>

#define D_MODEL 1024
#define NUM_HEADS 16
#define DEPTH 64
#define BATCH 2
#define SEQ 2048
#define MTOK (BATCH * SEQ)         // 4096 token rows
#define BHEADS (BATCH * NUM_HEADS) // 32

typedef __attribute__((ext_vector_type(16))) __bf16 bf16x16;
typedef __attribute__((ext_vector_type(8)))  float  floatx8;

// ---------------------------------------------------------------------------
// WMMA fragment loaders (wave32, v_wmma_f32_16x16x32_bf16 layouts per ISA)
//
// A 16x32 bf16:  lane = (kHalf<<4)|m ; elems 0..7  -> K = kHalf*8 + i
//                                      elems 8..15 -> K = 16 + kHalf*8 + (i-8)
// B 32x16 bf16 (N-major source, Bt[n][k]): lane = (kHalf<<4)|n ;
//                                      elems 0..7  -> K = kHalf*8 + i
//                                      elems 8..15 -> K = 16 + kHalf*8 + (i-8)
// C/D 16x16 f32: n = lane&15, m = vgpr + 8*(lane>>4)
// ---------------------------------------------------------------------------

__device__ __forceinline__ bf16x16 load_afrag(const __bf16* __restrict__ p,
                                              int ld, int k0, int lane) {
  const int r  = lane & 15;
  const int kh = (lane >> 4) << 3;  // 0 or 8
  const __bf16* p0 = p + (size_t)r * ld + k0 + kh;
  union { bf16x16 v; uint4 q[2]; } f;
  f.q[0] = *(const uint4*)(p0);
  f.q[1] = *(const uint4*)(p0 + 16);
  return f.v;
}

__device__ __forceinline__ bf16x16 load_afrag(const float* __restrict__ p,
                                              int ld, int k0, int lane) {
  const int r  = lane & 15;
  const int kh = (lane >> 4) << 3;
  const float* p0 = p + (size_t)r * ld + k0 + kh;
  float4 a0 = *(const float4*)(p0);
  float4 a1 = *(const float4*)(p0 + 4);
  float4 b0 = *(const float4*)(p0 + 16);
  float4 b1 = *(const float4*)(p0 + 20);
  bf16x16 v;
  v[0]  = (__bf16)a0.x; v[1]  = (__bf16)a0.y; v[2]  = (__bf16)a0.z; v[3]  = (__bf16)a0.w;
  v[4]  = (__bf16)a1.x; v[5]  = (__bf16)a1.y; v[6]  = (__bf16)a1.z; v[7]  = (__bf16)a1.w;
  v[8]  = (__bf16)b0.x; v[9]  = (__bf16)b0.y; v[10] = (__bf16)b0.z; v[11] = (__bf16)b0.w;
  v[12] = (__bf16)b1.x; v[13] = (__bf16)b1.y; v[14] = (__bf16)b1.z; v[15] = (__bf16)b1.w;
  return v;
}

__device__ __forceinline__ bf16x16 load_bfrag(const __bf16* __restrict__ p,
                                              int ld, int k0, int lane) {
  const int n  = lane & 15;
  const int kh = (lane >> 4) << 3;  // 0 or 8
  const __bf16* p0 = p + (size_t)n * ld + k0 + kh;
  union { bf16x16 v; uint4 q[2]; } f;
  f.q[0] = *(const uint4*)(p0);
  f.q[1] = *(const uint4*)(p0 + 16);
  return f.v;
}

__device__ __forceinline__ floatx8 wmma_bf16(bf16x16 a, bf16x16 b, floatx8 c) {
  return __builtin_amdgcn_wmma_f32_16x16x32_bf16(false, a, false, b,
                                                 (short)0, c, false, false);
}

__device__ __forceinline__ floatx8 zero8() {
  floatx8 z = {0.f, 0.f, 0.f, 0.f, 0.f, 0.f, 0.f, 0.f};
  return z;
}

__device__ __forceinline__ void zero_acc(floatx8 acc[2][4]) {
#pragma unroll
  for (int mt = 0; mt < 2; ++mt)
#pragma unroll
    for (int t = 0; t < 4; ++t) acc[mt][t] = zero8();
}

// ---------------------------------------------------------------------------
// Shared wave-level GEMM core: one wave computes a 32(M) x 64(N) tile.
// ---------------------------------------------------------------------------
template <typename AT>
__device__ __forceinline__ void gemm_32x64(const AT* __restrict__ A, int lda,
                                           const __bf16* __restrict__ Bt, int ldb,
                                           int Ktot, int lane,
                                           floatx8 acc[2][4]) {
  for (int k0 = 0; k0 < Ktot; k0 += 32) {
    __builtin_prefetch(A + (size_t)(lane & 15) * lda + k0 + 32, 0, 1);
    bf16x16 a0 = load_afrag(A, lda, k0, lane);
    bf16x16 a1 = load_afrag(A + (size_t)16 * lda, lda, k0, lane);
    bf16x16 b[4];
#pragma unroll
    for (int t = 0; t < 4; ++t)
      b[t] = load_bfrag(Bt + (size_t)(t * 16) * ldb, ldb, k0, lane);
#pragma unroll
    for (int t = 0; t < 4; ++t) {
      acc[0][t] = wmma_bf16(a0, b[t], acc[0][t]);
      acc[1][t] = wmma_bf16(a1, b[t], acc[1][t]);
    }
  }
}

// ---------------------------------------------------------------------------
// 1) Weight convert + transpose:  W[k][n] f32  ->  WT[n][k] bf16
// ---------------------------------------------------------------------------
__global__ void wtrans_kernel(const float* __restrict__ W, __bf16* __restrict__ WT) {
  const int idx = blockIdx.x * 256 + threadIdx.x;  // idx = n*1024 + k
  const int n = idx >> 10;
  const int k = idx & (D_MODEL - 1);
  WT[idx] = (__bf16)W[(size_t)k * D_MODEL + n];
}

// ---------------------------------------------------------------------------
// 2) Projection GEMM: Y = X @ W + b  (layout selected at compile time)
//    TRANS=0: dst[b,h,s,d] bf16 (q/k).  TRANS=1: dst[b,h,d,s] bf16 (v),
//    with the 8 consecutive s-positions packed into one b128 store.
//    block=128 (4 waves), wave -> 32(M) x 64(N); grid (4096/128, 1024/64)
// ---------------------------------------------------------------------------
template <int TRANS>
__global__ void proj_kernel(const float* __restrict__ X,
                            const __bf16* __restrict__ WT,
                            const float* __restrict__ bias,
                            __bf16* __restrict__ dst) {
  const int lane = threadIdx.x & 31;
  const int wave = threadIdx.x >> 5;
  const int m0 = (blockIdx.x * 4 + wave) * 32;
  const int n0 = blockIdx.y * 64;

  floatx8 acc[2][4];
  zero_acc(acc);
  gemm_32x64(X + (size_t)m0 * D_MODEL, D_MODEL,
             WT + (size_t)n0 * D_MODEL, D_MODEL, D_MODEL, lane, acc);

  const int mh = (lane >> 4) << 3;
#pragma unroll
  for (int mt = 0; mt < 2; ++mt) {
    const int mbase = m0 + mt * 16 + mh;      // 8 consecutive token rows
    const int b = mbase >> 11;                // / SEQ (constant over i)
    const int s0 = mbase & (SEQ - 1);
#pragma unroll
    for (int t = 0; t < 4; ++t) {
      const int n = n0 + t * 16 + (lane & 15);
      const int h = n >> 6;
      const int d = n & (DEPTH - 1);
      const float bv = bias[n];
      if (TRANS) {
        // 8 consecutive s positions -> one 16B store
        union { __bf16 e[8]; uint4 v; } pk;
#pragma unroll
        for (int i = 0; i < 8; ++i) pk.e[i] = (__bf16)(acc[mt][t][i] + bv);
        __bf16* p = dst + (((size_t)(b * NUM_HEADS + h)) * DEPTH + d) * SEQ + s0;
        *(uint4*)p = pk.v;
      } else {
#pragma unroll
        for (int i = 0; i < 8; ++i) {
          const float val = acc[mt][t][i] + bv;
          dst[(((size_t)(b * NUM_HEADS + h)) * SEQ + (s0 + i)) * DEPTH + d] =
              (__bf16)val;
        }
      }
    }
  }
}

// ---------------------------------------------------------------------------
// 3) Fused scores + softmax + attn@V.
//    One block (4 waves) owns 32 query rows of one head; the full 32x2048
//    f32 score strip lives in LDS (256KB of the WGP's 320KB).
//      phase 1: scores -> LDS (Q frags cached in regs; waves split keys)
//      phase 2: softmax per row (wave/row, __shfl_xor reductions); write
//               normalized probs to d_out attn AND back to LDS
//      phase 3: ctx = probs(LDS) @ V (waves split K, LDS partial reduce)
//    grid (SEQ/32, 1, BHEADS), block=128, dynamic LDS = 288KB
// ---------------------------------------------------------------------------
__global__ void fused_attn_kernel(const __bf16* __restrict__ q,
                                  const __bf16* __restrict__ k,
                                  const __bf16* __restrict__ vT,
                                  const float* __restrict__ mask,
                                  float* __restrict__ attn,
                                  __bf16* __restrict__ ctx) {
  extern __shared__ float smem[];
  float* strip = smem;               // [32][2048] f32 scores/probs
  float* redws = smem + 32 * SEQ;    // [4][32*64] f32 partial ctx

  const int lane = threadIdx.x & 31;
  const int wave = threadIdx.x >> 5;
  const int bh = blockIdx.z;
  const int b = bh >> 4;
  const int h = bh & (NUM_HEADS - 1);
  const int row0 = blockIdx.x * 32;

  const __bf16* qh = q + (size_t)bh * SEQ * DEPTH + (size_t)row0 * DEPTH;
  const __bf16* kh = k + (size_t)bh * SEQ * DEPTH;
  const __bf16* vh = vT + (size_t)bh * DEPTH * SEQ;
  float* attnh = attn + (size_t)bh * SEQ * SEQ + (size_t)row0 * SEQ;

  const int mh = (lane >> 4) << 3;

  // ---- phase 1: scores -> LDS ------------------------------------------
  bf16x16 aq[2][2];
#pragma unroll
  for (int ks = 0; ks < 2; ++ks)
#pragma unroll
    for (int mt = 0; mt < 2; ++mt)
      aq[ks][mt] = load_afrag(qh + (size_t)(mt * 16) * DEPTH, DEPTH, ks * 32, lane);

  for (int nt = 0; nt < 8; ++nt) {           // this wave's 8 key tiles of 64
    const int n0 = wave * 512 + nt * 64;
    floatx8 acc[2][4];
    zero_acc(acc);
#pragma unroll
    for (int ks = 0; ks < 2; ++ks) {
      bf16x16 bfr[4];
#pragma unroll
      for (int t = 0; t < 4; ++t)
        bfr[t] = load_bfrag(kh + (size_t)(n0 + t * 16) * DEPTH, DEPTH, ks * 32, lane);
#pragma unroll
      for (int t = 0; t < 4; ++t) {
        acc[0][t] = wmma_bf16(aq[ks][0], bfr[t], acc[0][t]);
        acc[1][t] = wmma_bf16(aq[ks][1], bfr[t], acc[1][t]);
      }
    }
#pragma unroll
    for (int mt = 0; mt < 2; ++mt) {
#pragma unroll
      for (int t = 0; t < 4; ++t) {
        const int n = n0 + t * 16 + (lane & 15);
        const float mv = mask[b * SEQ + n] * -1e9f;
#pragma unroll
        for (int i = 0; i < 8; ++i) {
          const int m = mt * 16 + mh + i;
          strip[m * SEQ + n] = acc[mt][t][i] * 0.125f + mv;
        }
      }
    }
  }
  __syncthreads();

  // ---- phase 2: softmax per row (wave handles 8 rows) -------------------
  for (int rr = 0; rr < 8; ++rr) {
    const int r = wave * 8 + rr;
    float* row = strip + r * SEQ;

    float mx = -3.0e38f;
#pragma unroll 8
    for (int j = 0; j < 64; ++j) mx = fmaxf(mx, row[lane + j * 32]);
#pragma unroll
    for (int off = 16; off > 0; off >>= 1) mx = fmaxf(mx, __shfl_xor(mx, off, 32));

    float sum = 0.f;
#pragma unroll 8
    for (int j = 0; j < 64; ++j) sum += __expf(row[lane + j * 32] - mx);
#pragma unroll
    for (int off = 16; off > 0; off >>= 1) sum += __shfl_xor(sum, off, 32);
    const float inv = 1.0f / sum;

    float* grow = attnh + (size_t)r * SEQ;
#pragma unroll 8
    for (int j = 0; j < 64; ++j) {
      const int c = lane + j * 32;
      const float e = __expf(row[c] - mx) * inv;
      grow[c] = e;       // attention output (HBM, written once)
      row[c] = e;        // normalized probs kept in LDS for phase 3
    }
  }
  __syncthreads();

  // ---- phase 3: ctx = probs @ V, waves split K --------------------------
  floatx8 acc[2][4];
  zero_acc(acc);
  const float* Astrip = strip;  // [32][2048]
  for (int kk = wave * 512; kk < wave * 512 + 512; kk += 32) {
    bf16x16 a0 = load_afrag(Astrip, SEQ, kk, lane);
    bf16x16 a1 = load_afrag(Astrip + 16 * SEQ, SEQ, kk, lane);
    bf16x16 bfr[4];
#pragma unroll
    for (int t = 0; t < 4; ++t)
      bfr[t] = load_bfrag(vh + (size_t)(t * 16) * SEQ, SEQ, kk, lane);
#pragma unroll
    for (int t = 0; t < 4; ++t) {
      acc[0][t] = wmma_bf16(a0, bfr[t], acc[0][t]);
      acc[1][t] = wmma_bf16(a1, bfr[t], acc[1][t]);
    }
  }
  // dump per-wave partials
#pragma unroll
  for (int mt = 0; mt < 2; ++mt)
#pragma unroll
    for (int t = 0; t < 4; ++t)
#pragma unroll
      for (int i = 0; i < 8; ++i) {
        const int m = mt * 16 + mh + i;
        redws[wave * 2048 + m * 64 + t * 16 + (lane & 15)] = acc[mt][t][i];
      }
  __syncthreads();

  // reduce 4 partials and store ctx (bf16)
  for (int e = threadIdx.x; e < 2048; e += 128) {
    const float s =
        redws[e] + redws[2048 + e] + redws[4096 + e] + redws[6144 + e];
    const int m = e >> 6;
    const int d = e & 63;
    ctx[((size_t)(b * SEQ + row0 + m)) * D_MODEL + h * DEPTH + d] = (__bf16)s;
  }
}

// ---------------------------------------------------------------------------
// 4) out = ctx @ Wo + bo, f32 output. grid (4096/128, 1024/64), block=128
// ---------------------------------------------------------------------------
__global__ void outproj_kernel(const __bf16* __restrict__ ctx,
                               const __bf16* __restrict__ WoT,
                               const float* __restrict__ bo,
                               float* __restrict__ out) {
  const int lane = threadIdx.x & 31;
  const int wave = threadIdx.x >> 5;
  const int m0 = (blockIdx.x * 4 + wave) * 32;
  const int n0 = blockIdx.y * 64;

  floatx8 acc[2][4];
  zero_acc(acc);
  gemm_32x64(ctx + (size_t)m0 * D_MODEL, D_MODEL,
             WoT + (size_t)n0 * D_MODEL, D_MODEL, D_MODEL, lane, acc);

  const int mh = (lane >> 4) << 3;
#pragma unroll
  for (int mt = 0; mt < 2; ++mt) {
#pragma unroll
    for (int t = 0; t < 4; ++t) {
      const int n = n0 + t * 16 + (lane & 15);
      const float bv = bo[n];
#pragma unroll
      for (int i = 0; i < 8; ++i) {
        const int m = m0 + mt * 16 + mh + i;
        out[(size_t)m * D_MODEL + n] = acc[mt][t][i] + bv;
      }
    }
  }
}

// ---------------------------------------------------------------------------
extern "C" void kernel_launch(void* const* d_in, const int* in_sizes, int n_in,
                              void* d_out, int out_size, void* d_ws, size_t ws_size,
                              hipStream_t stream) {
  (void)in_sizes; (void)n_in; (void)out_size; (void)ws_size;

  const float* Q    = (const float*)d_in[0];
  const float* K    = (const float*)d_in[1];
  const float* V    = (const float*)d_in[2];
  const float* mask = (const float*)d_in[3];
  const float* Wq   = (const float*)d_in[4];
  const float* bq   = (const float*)d_in[5];
  const float* Wk   = (const float*)d_in[6];
  const float* bk   = (const float*)d_in[7];
  const float* Wv   = (const float*)d_in[8];
  const float* bv   = (const float*)d_in[9];
  const float* Wo   = (const float*)d_in[10];
  const float* bo   = (const float*)d_in[11];

  float* out  = (float*)d_out;                       // [4096,1024] f32
  float* attn = out + (size_t)MTOK * D_MODEL;        // [32,2048,2048] f32

  // workspace layout (bf16 elements)
  __bf16* ws   = (__bf16*)d_ws;
  const size_t WSZ = (size_t)D_MODEL * D_MODEL;      // 1M
  const size_t PSZ = (size_t)BHEADS * SEQ * DEPTH;   // 4M
  __bf16* wqT = ws;
  __bf16* wkT = wqT + WSZ;
  __bf16* wvT = wkT + WSZ;
  __bf16* woT = wvT + WSZ;
  __bf16* qb  = woT + WSZ;   // [b,h,s,d]
  __bf16* kb  = qb + PSZ;    // [b,h,s,d]
  __bf16* vTb = kb + PSZ;    // [b,h,d,s]
  __bf16* ctx = vTb + PSZ;   // [4096,1024]

  const dim3 blk128(128, 1, 1);
  const dim3 wgrid((D_MODEL * D_MODEL) / 256, 1, 1);

  // 1) weights -> bf16 transposed
  wtrans_kernel<<<wgrid, 256, 0, stream>>>(Wq, wqT);
  wtrans_kernel<<<wgrid, 256, 0, stream>>>(Wk, wkT);
  wtrans_kernel<<<wgrid, 256, 0, stream>>>(Wv, wvT);
  wtrans_kernel<<<wgrid, 256, 0, stream>>>(Wo, woT);

  // 2) projections
  const dim3 pgrid(MTOK / 128, D_MODEL / 64, 1);
  proj_kernel<0><<<pgrid, blk128, 0, stream>>>(Q, wqT, bq, qb);
  proj_kernel<0><<<pgrid, blk128, 0, stream>>>(K, wkT, bk, kb);
  proj_kernel<1><<<pgrid, blk128, 0, stream>>>(V, wvT, bv, vTb);

  // 3) fused scores + softmax + attn@V (attn written once to d_out)
  const size_t lds_bytes = (size_t)(32 * SEQ + 4 * 2048) * sizeof(float); // 288KB
  const dim3 fgrid(SEQ / 32, 1, BHEADS);
  fused_attn_kernel<<<fgrid, blk128, lds_bytes, stream>>>(qb, kb, vTb, mask,
                                                          attn, ctx);

  // 4) output projection -> out (f32)
  const dim3 ogrid(MTOK / 128, D_MODEL / 64, 1);
  outproj_kernel<<<ogrid, blk128, 0, stream>>>(ctx, woT, bo, out);
}
